// RewardGuidanceModel_64725157150922
// MI455X (gfx1250) — compile-verified, matrix operand order checked by reading, not run on hardware
//
#include <hip/hip_runtime.h>
#include <hip/hip_bf16.h>
#include <math.h>

// ---------------------------------------------------------------------------
// Model constants
#define BD      512            // batch
#define LSEQ    17             // sequence length
#define MTOK    (BD*LSEQ)      // 8704 token rows (= 544 tiles of 16)
#define DM      128            // d_model
#define DINNER  256
#define DSTATE  32
#define DCONV   4
#define NHEADS  32
#define HEADDIM 8
#define CONVD   320            // D_INNER + 2*D_STATE
#define DPROJ   608            // 2*D_INNER + 2*D_STATE + NHEADS
#define NLAYER  8
#define NBIN    392
#define NBOUT   32

typedef __bf16 bf16;
typedef __attribute__((ext_vector_type(4)))  __bf16 v4bf;
typedef __attribute__((ext_vector_type(8)))  __bf16 v8bf;
typedef __attribute__((ext_vector_type(16))) __bf16 v16bf;
typedef __attribute__((ext_vector_type(8)))  float  v8f;

// ---------------------------------------------------------------------------
// WMMA helpers (CDNA5 wave32, 16x16x32 bf16 -> f32)
__device__ __forceinline__ v8f wmma_bf16(v16bf a, v16bf b, v8f c) {
  return __builtin_amdgcn_wmma_f32_16x16x32_bf16(false, a, false, b,
                                                 (short)0, c, false, false);
}

// Load one A/B fragment row for the 16-bit 16x32 layout:
// lane L holds row (L&15); elements 0..7 = K = kb + 8*(L>>4) .. +7,
// elements 8..15 = K = kb + 16 + 8*(L>>4) .. +7.
__device__ __forceinline__ v16bf load_frag(const bf16* __restrict__ row,
                                           int kb, int half) {
  v8bf c0 = *(const v8bf*)(row + kb + 8 * half);
  v8bf c1 = *(const v8bf*)(row + kb + 16 + 8 * half);
  return __builtin_shufflevector(c0, c1, 0,1,2,3,4,5,6,7,8,9,10,11,12,13,14,15);
}

__device__ __forceinline__ float sigmoidf_(float x) { return 1.f / (1.f + expf(-x)); }

// Async global->LDS copy, 16 bytes per active lane (CDNA5, ASYNCcnt-tracked)
__device__ __forceinline__ void async_copy_b128(unsigned lds_addr,
                                                const float* gaddr) {
  asm volatile("global_load_async_to_lds_b128 %0, %1, off"
               :: "v"(lds_addr), "v"(gaddr) : "memory");
}
__device__ __forceinline__ void wait_async0() {
  asm volatile("s_wait_asynccnt 0x0" ::: "memory");
}

// ---------------------------------------------------------------------------
// tfe[b][j] = time_flags[b]*tf_W[j] + tf_b[j]
__global__ void k_tfe(const float* __restrict__ tflags, const float* __restrict__ tfW,
                      const float* __restrict__ tfb, float* __restrict__ tfe) {
  int idx = blockIdx.x * 256 + threadIdx.x;         // BD*512
  int b = idx >> 9, j = idx & 511;
  tfe[idx] = tflags[b] * tfW[j] + tfb[j];
}

// Split fp32 weights into transposed bf16 hi/lo: W[l][K][N] -> T[l][N][K]
__global__ void k_split_w(const float* __restrict__ W, bf16* __restrict__ Th,
                          bf16* __restrict__ Tl, int K, int N) {
  int idx = blockIdx.x * 256 + threadIdx.x;         // NLAYER*K*N, contiguous in K
  int l = idx / (K * N);
  int rem = idx - l * (K * N);
  int n = rem / K, k = rem - n * K;
  float v = W[(l * K + k) * N + n];
  bf16 h = (bf16)v;
  Th[idx] = h;
  Tl[idx] = (bf16)(v - (float)h);
}

// Embedding + input projection + tfe scale/shift: builds x (MTOK x 128) fp32
__global__ void k_embed(const float* __restrict__ init_s, const float* __restrict__ fut_s,
                        const float* __restrict__ semb, const float* __restrict__ inW,
                        const float* __restrict__ inb, const float* __restrict__ tfe,
                        float* __restrict__ x) {
  int idx = blockIdx.x * 256 + threadIdx.x;         // MTOK*128
  int row = idx >> 7, d = idx & 127;
  int b = row / LSEQ, l = row - b * LSEQ;
  float v;
  if (d < 64) {
    v = semb[l * 64 + d];
  } else {
    int j = d - 64;
    const float* src = (l == 0) ? (init_s + b * NBIN)
                                : (fut_s + ((size_t)b * 16 + (l - 1)) * NBIN);
    float acc = inb[j];
    for (int k = 0; k < NBIN; ++k) acc += src[k] * inW[k * 64 + j];
    v = acc;
  }
  x[idx] = v * tfe[b * 512 + d] + tfe[b * 512 + 128 + d];
}

// Per-row RMSNorm over 128 + bf16 hi/lo split. 1 wave per row, 8 rows/block.
__global__ void k_rmsnorm_split(const float* __restrict__ X, const float* __restrict__ nw,
                                bf16* __restrict__ Xh, bf16* __restrict__ Xl) {
  int wid = threadIdx.x >> 5, lane = threadIdx.x & 31;
  int row = blockIdx.x * 8 + wid;
  float4 v = ((const float4*)(X + (size_t)row * 128))[lane];
  float ss = v.x * v.x + v.y * v.y + v.z * v.z + v.w * v.w;
  #pragma unroll
  for (int o = 16; o; o >>= 1) ss += __shfl_xor(ss, o, 32);
  float sc = rsqrtf(ss * (1.f / 128.f) + 1e-5f);
  float4 w = ((const float4*)nw)[lane];
  float ov[4] = {v.x * sc * w.x, v.y * sc * w.y, v.z * sc * w.z, v.w * sc * w.w};
  v4bf hv, lv;
  #pragma unroll
  for (int j = 0; j < 4; ++j) {
    bf16 hb = (bf16)ov[j];
    hv[j] = hb;
    lv[j] = (bf16)(ov[j] - (float)hb);
  }
  ((v4bf*)(Xh + (size_t)row * 128))[lane] = hv;
  ((v4bf*)(Xl + (size_t)row * 128))[lane] = lv;
}

// Inproj GEMM: zx (MTOK x 608) = Xn (MTOK x 128) @ W (128 x 608), split-bf16.
// 1 wave = one 16x32 output strip (2 N-tiles share the A fragments).
// 544 M-tiles x 19 N-groups, 8 waves/block.
__global__ void k_gemm_in(const bf16* __restrict__ Ah, const bf16* __restrict__ Al,
                          const bf16* __restrict__ Bh, const bf16* __restrict__ Bl,
                          float* __restrict__ Y) {
  int wave = threadIdx.x >> 5, lane = threadIdx.x & 31;
  int wtile = blockIdx.x * 8 + wave;
  const int NG = DPROJ / 32;                        // 19
  int mt = wtile / NG, ng = wtile - mt * NG;
  int r = lane & 15, half = lane >> 4;
  const bf16* arh = Ah + (size_t)(mt * 16 + r) * 128;
  const bf16* arl = Al + (size_t)(mt * 16 + r) * 128;
  const bf16* b0h = Bh + (size_t)(ng * 32 + r) * 128;
  const bf16* b0l = Bl + (size_t)(ng * 32 + r) * 128;
  const bf16* b1h = b0h + (size_t)16 * 128;
  const bf16* b1l = b0l + (size_t)16 * 128;
  v8f acc0 = {}, acc1 = {};
  #pragma unroll
  for (int kb = 0; kb < 128; kb += 32) {
    v16bf ah  = load_frag(arh, kb, half);
    v16bf al  = load_frag(arl, kb, half);
    v16bf bh0 = load_frag(b0h, kb, half);
    v16bf bl0 = load_frag(b0l, kb, half);
    v16bf bh1 = load_frag(b1h, kb, half);
    v16bf bl1 = load_frag(b1l, kb, half);
    acc0 = wmma_bf16(ah, bh0, acc0);
    acc0 = wmma_bf16(ah, bl0, acc0);
    acc0 = wmma_bf16(al, bh0, acc0);
    acc1 = wmma_bf16(ah, bh1, acc1);
    acc1 = wmma_bf16(ah, bl1, acc1);
    acc1 = wmma_bf16(al, bh1, acc1);
  }
  // C/D layout: element e -> row m0 + e + 8*half, col n0 + (lane&15)
  float* yp = Y + (size_t)(mt * 16 + 8 * half) * DPROJ + ng * 32 + r;
  #pragma unroll
  for (int e = 0; e < 8; ++e) {
    yp[(size_t)e * DPROJ]      = acc0[e];
    yp[(size_t)e * DPROJ + 16] = acc1[e];
  }
}

// Outproj GEMM with fused residual: X (MTOK x 128) += Yn (MTOK x 256) @ W(256x128)
// 1 wave = 16x32 strip; 544 M-tiles x 4 N-groups.
__global__ void k_gemm_out(const bf16* __restrict__ Ah, const bf16* __restrict__ Al,
                           const bf16* __restrict__ Bh, const bf16* __restrict__ Bl,
                           float* __restrict__ X) {
  int wave = threadIdx.x >> 5, lane = threadIdx.x & 31;
  int wtile = blockIdx.x * 8 + wave;
  int mt = wtile >> 2, ng = wtile & 3;              // N = 128 -> 4 groups of 32
  int r = lane & 15, half = lane >> 4;
  const bf16* arh = Ah + (size_t)(mt * 16 + r) * 256;
  const bf16* arl = Al + (size_t)(mt * 16 + r) * 256;
  const bf16* b0h = Bh + (size_t)(ng * 32 + r) * 256;
  const bf16* b0l = Bl + (size_t)(ng * 32 + r) * 256;
  const bf16* b1h = b0h + (size_t)16 * 256;
  const bf16* b1l = b0l + (size_t)16 * 256;
  v8f acc0 = {}, acc1 = {};
  #pragma unroll
  for (int kb = 0; kb < 256; kb += 32) {
    v16bf ah  = load_frag(arh, kb, half);
    v16bf al  = load_frag(arl, kb, half);
    v16bf bh0 = load_frag(b0h, kb, half);
    v16bf bl0 = load_frag(b0l, kb, half);
    v16bf bh1 = load_frag(b1h, kb, half);
    v16bf bl1 = load_frag(b1l, kb, half);
    acc0 = wmma_bf16(ah, bh0, acc0);
    acc0 = wmma_bf16(ah, bl0, acc0);
    acc0 = wmma_bf16(al, bh0, acc0);
    acc1 = wmma_bf16(ah, bh1, acc1);
    acc1 = wmma_bf16(ah, bl1, acc1);
    acc1 = wmma_bf16(al, bh1, acc1);
  }
  float* xp = X + (size_t)(mt * 16 + 8 * half) * 128 + ng * 32 + r;
  #pragma unroll
  for (int e = 0; e < 8; ++e) {
    xp[(size_t)e * 128]      += acc0[e];
    xp[(size_t)e * 128 + 16] += acc1[e];
  }
}

// Depthwise conv + SiLU + SSM scan + gating + group RMSNorm.
// 1 block per batch, 256 threads = (head h = tid>>3, p = tid&7).
// Raw xBC slab staged to LDS with async b128 copies (ASYNCcnt), overlapped
// with the dt/dA transcendental work, drained with s_wait_asynccnt.
__global__ void k_conv_scan(const float* __restrict__ zx, const float* __restrict__ cw,
                            const float* __restrict__ cb, const float* __restrict__ dtb,
                            const float* __restrict__ Alog, const float* __restrict__ Dp,
                            const float* __restrict__ gw,
                            bf16* __restrict__ Yh, bf16* __restrict__ Yl) {
  __shared__ float s_raw[LSEQ * CONVD];             // raw xBC (pre-conv)
  __shared__ float s_xbc[LSEQ * CONVD];             // conv+SiLU output
  __shared__ float s_dt[LSEQ * NHEADS];
  __shared__ float s_dA[LSEQ * NHEADS];
  __shared__ float s_red[8];
  int tid = threadIdx.x;
  int b = blockIdx.x;
  int row0 = b * LSEQ;

  // 1) issue async copies of the raw 17x320 slab (80 b128 chunks per row)
  for (int idx = tid; idx < (LSEQ * CONVD) / 4; idx += 256) {
    int l = idx / (CONVD / 4);
    int c4 = idx - l * (CONVD / 4);
    const float* g = zx + (size_t)(row0 + l) * DPROJ + DINNER + c4 * 4;
    async_copy_b128((unsigned)(uintptr_t)&s_raw[idx * 4], g);
  }

  // 2) dt = softplus(raw + bias); dA = exp(dt * (-exp(A_log)))  (overlaps copies)
  for (int idx = tid; idx < LSEQ * NHEADS; idx += 256) {
    int l = idx >> 5, h = idx & 31;
    float raw = zx[(size_t)(row0 + l) * DPROJ + (DINNER + CONVD) + h] + dtb[h];
    float dt = (raw > 20.f) ? raw : log1pf(expf(raw));
    s_dt[idx] = dt;
    s_dA[idx] = expf(dt * (-expf(Alog[h])));
  }
  wait_async0();
  __syncthreads();

  // 3) causal depthwise conv (K=4) + SiLU from LDS
  for (int idx = tid; idx < LSEQ * CONVD; idx += 256) {
    int l = idx / CONVD, c = idx - l * CONVD;
    float acc = cb[c];
    #pragma unroll
    for (int k = 0; k < DCONV; ++k) {
      int t = l + k - (DCONV - 1);
      float xv = (t >= 0) ? s_raw[t * CONVD + c] : 0.f;
      acc += cw[c * DCONV + k] * xv;
    }
    s_xbc[idx] = acc * sigmoidf_(acc);
  }
  __syncthreads();

  // 4) sequential SSM scan with register-resident 32-deep state
  int h = tid >> 3;
  int lane = tid & 31, wid = tid >> 5;
  float st[DSTATE];
  #pragma unroll
  for (int n = 0; n < DSTATE; ++n) st[n] = 0.f;
  float dpar = Dp[h];

  #pragma unroll
  for (int l = 0; l < LSEQ; ++l) {
    float dt = s_dt[l * NHEADS + h];
    float dA = s_dA[l * NHEADS + h];
    float xv = s_xbc[l * CONVD + tid];
    float coeff = dt * xv;
    float y = 0.f;
    #pragma unroll
    for (int n = 0; n < DSTATE; ++n) {
      st[n] = st[n] * dA + coeff * s_xbc[l * CONVD + DINNER + n];          // B
      y += st[n] * s_xbc[l * CONVD + DINNER + DSTATE + n];                 // C
    }
    y += dpar * xv;
    // gating with z, then RMSNorm over 256 dims
    float z = zx[(size_t)(row0 + l) * DPROJ + tid];
    float g = y * (z * sigmoidf_(z));
    float ss = g * g;
    #pragma unroll
    for (int o = 16; o; o >>= 1) ss += __shfl_xor(ss, o, 32);
    if (lane == 0) s_red[wid] = ss;
    __syncthreads();
    float tot = s_red[0] + s_red[1] + s_red[2] + s_red[3] +
                s_red[4] + s_red[5] + s_red[6] + s_red[7];
    float sc = rsqrtf(tot * (1.f / 256.f) + 1e-5f);
    float yn = g * sc * gw[tid];
    bf16 hb = (bf16)yn;
    Yh[(size_t)(row0 + l) * DINNER + tid] = hb;
    Yl[(size_t)(row0 + l) * DINNER + tid] = (bf16)(yn - (float)hb);
    __syncthreads();
  }
}

// Final norm + tfe scale/shift + 128->32 projection. 1 block (128 thr) per row.
__global__ void k_final(const float* __restrict__ x, const float* __restrict__ fnw,
                        const float* __restrict__ tfe, const float* __restrict__ outW,
                        const float* __restrict__ outb, float* __restrict__ out) {
  __shared__ float s_v[128];
  __shared__ float s_red[4];
  int row = blockIdx.x;
  int b = row / LSEQ;
  int tid = threadIdx.x;
  float xv = x[(size_t)row * 128 + tid];
  float ss = xv * xv;
  #pragma unroll
  for (int o = 16; o; o >>= 1) ss += __shfl_xor(ss, o, 32);
  int lane = tid & 31, wid = tid >> 5;
  if (lane == 0) s_red[wid] = ss;
  __syncthreads();
  float tot = s_red[0] + s_red[1] + s_red[2] + s_red[3];
  float sc = rsqrtf(tot * (1.f / 128.f) + 1e-5f);
  float v = xv * sc * fnw[tid];
  v = v * tfe[b * 512 + 256 + tid] + tfe[b * 512 + 384 + tid];
  s_v[tid] = v;
  __syncthreads();
  if (tid < NBOUT) {
    float acc = outb[tid];
    #pragma unroll 8
    for (int d = 0; d < 128; ++d) acc += s_v[d] * outW[d * NBOUT + tid];
    out[(size_t)row * NBOUT + tid] = acc;
  }
}

// ---------------------------------------------------------------------------
extern "C" void kernel_launch(void* const* d_in, const int* in_sizes, int n_in,
                              void* d_out, int out_size, void* d_ws, size_t ws_size,
                              hipStream_t stream) {
  const float* init_states   = (const float*)d_in[0];
  const float* future_states = (const float*)d_in[1];
  const float* time_flags    = (const float*)d_in[2];
  const float* states_emb    = (const float*)d_in[3];
  const float* in_W          = (const float*)d_in[4];
  const float* in_b          = (const float*)d_in[5];
  const float* tf_W          = (const float*)d_in[6];
  const float* tf_b          = (const float*)d_in[7];
  const float* out_W         = (const float*)d_in[8];
  const float* out_b         = (const float*)d_in[9];
  const float* norm_w        = (const float*)d_in[10];
  const float* inproj_W      = (const float*)d_in[11];
  const float* conv_w        = (const float*)d_in[12];
  const float* conv_b        = (const float*)d_in[13];
  const float* dt_bias       = (const float*)d_in[14];
  const float* A_log         = (const float*)d_in[15];
  const float* D_param       = (const float*)d_in[16];
  const float* gnorm_w       = (const float*)d_in[17];
  const float* outproj_W     = (const float*)d_in[18];
  const float* final_norm_w  = (const float*)d_in[19];
  float* out = (float*)d_out;

  char* w = (char*)d_ws;
  size_t off = 0;
  auto alloc = [&](size_t bytes) -> void* {
    off = (off + 255) & ~(size_t)255;
    void* p = w + off;
    off += bytes;
    return p;
  };
  float* tfe  = (float*)alloc((size_t)BD * 512 * 4);
  float* x    = (float*)alloc((size_t)MTOK * DM * 4);
  bf16*  Xh   = (bf16*) alloc((size_t)MTOK * DM * 2);
  bf16*  Xl   = (bf16*) alloc((size_t)MTOK * DM * 2);
  float* zx   = (float*)alloc((size_t)MTOK * DPROJ * 4);
  bf16*  Yh   = (bf16*) alloc((size_t)MTOK * DINNER * 2);
  bf16*  Yl   = (bf16*) alloc((size_t)MTOK * DINNER * 2);
  bf16*  WIh  = (bf16*) alloc((size_t)NLAYER * DPROJ * DM * 2);
  bf16*  WIl  = (bf16*) alloc((size_t)NLAYER * DPROJ * DM * 2);
  bf16*  WOh  = (bf16*) alloc((size_t)NLAYER * DM * DINNER * 2);
  bf16*  WOl  = (bf16*) alloc((size_t)NLAYER * DM * DINNER * 2);
  (void)ws_size; (void)n_in; (void)in_sizes; (void)out_size;

  // prep
  k_tfe<<<dim3((BD * 512) / 256), dim3(256), 0, stream>>>(time_flags, tf_W, tf_b, tfe);
  k_split_w<<<dim3((NLAYER * DM * DPROJ) / 256), dim3(256), 0, stream>>>(
      inproj_W, WIh, WIl, DM, DPROJ);
  k_split_w<<<dim3((NLAYER * DINNER * DM) / 256), dim3(256), 0, stream>>>(
      outproj_W, WOh, WOl, DINNER, DM);
  k_embed<<<dim3((MTOK * DM) / 256), dim3(256), 0, stream>>>(
      init_states, future_states, states_emb, in_W, in_b, tfe, x);

  // layers
  for (int i = 0; i < NLAYER; ++i) {
    k_rmsnorm_split<<<dim3(MTOK / 8), dim3(256), 0, stream>>>(
        x, norm_w + (size_t)i * DM, Xh, Xl);
    k_gemm_in<<<dim3((544 * (DPROJ / 32)) / 8), dim3(256), 0, stream>>>(
        Xh, Xl, WIh + (size_t)i * DPROJ * DM, WIl + (size_t)i * DPROJ * DM, zx);
    k_conv_scan<<<dim3(BD), dim3(256), 0, stream>>>(
        zx, conv_w + (size_t)i * CONVD * DCONV, conv_b + (size_t)i * CONVD,
        dt_bias + (size_t)i * NHEADS, A_log + (size_t)i * NHEADS,
        D_param + (size_t)i * NHEADS, gnorm_w + (size_t)i * DINNER, Yh, Yl);
    k_gemm_out<<<dim3((544 * 4) / 8), dim3(256), 0, stream>>>(
        Yh, Yl, WOh + (size_t)i * DM * DINNER, WOl + (size_t)i * DM * DINNER, x);
  }

  // epilogue
  k_final<<<dim3(MTOK), dim3(128), 0, stream>>>(
      x, final_norm_w, tfe, out_W, out_b, out);
}
